// TimeSeriesModel_1Step_ES_29463475650706
// MI455X (gfx1250) — matrix operand order repaired and checked by reading, not compiled, not run on hardware
//
#include <hip/hip_runtime.h>

// Problem constants (from reference): B=128, T=4096, I=64, H=128, G=3H=384, C=2
#define ALPHA_C 0.3f
#define BETA_C  0.5f
#define Bsz 128
#define Tsz 4096
#define Isz 64
#define Hsz 128
#define Gsz 384
#define Csz 2

typedef __attribute__((ext_vector_type(16))) _Float16 v16h;
typedef __attribute__((ext_vector_type(8)))  float    v8f;

union Frag16 {
    v16h v;
    unsigned int u[8];
    _Float16 h[16];
};

// ---------------------------------------------------------------------------
// Kernel 1: fused double-EMA over time, f32 -> f16 output for WMMA consumption.
// One thread per (b,i); x2h layout is [t][b*I + i] so each timestep's 128x64
// tile is contiguous. Captures st11 = x1[:,-1,1:3], st12 = x2[:,-1,1:3].
// ---------------------------------------------------------------------------
__global__ void ema_kernel(const float* __restrict__ x,
                           _Float16* __restrict__ x2h,
                           float* __restrict__ st11,
                           float* __restrict__ st12) {
    const int tid = blockIdx.x * blockDim.x + threadIdx.x;  // 0 .. B*I-1
    const int b = tid >> 6;
    const int i = tid & 63;
    const float* xp = x + (size_t)b * Tsz * Isz + i;

    float s1 = xp[0];
    float s2 = s1;
    x2h[tid] = (_Float16)s2;
    for (int t = 1; t < Tsz; ++t) {
        float xt = xp[(size_t)t * Isz];
        s1 = (1.0f - ALPHA_C) * xt + ALPHA_C * s1;
        s2 = (1.0f - BETA_C)  * s1 + BETA_C  * s2;
        x2h[(size_t)t * (Bsz * Isz) + tid] = (_Float16)s2;
    }
    if (i == 1 || i == 2) {
        st11[b * 2 + (i - 1)] = s1;
        st12[b * 2 + (i - 1)] = s2;
    }
}

// ---------------------------------------------------------------------------
// A-fragment loader (CDNA5 16-bit A layout: lane L: m=L&15, half=L>>4;
// 32-bit pair p holds k = kbase + ((p>=4)?16:0) + half*8 + (p&3)*2).
// ---------------------------------------------------------------------------
__device__ __forceinline__ v16h load_a_frag(const _Float16* row, int hs, int kbase) {
    Frag16 f;
    const unsigned int* r = (const unsigned int*)row;
#pragma unroll
    for (int p = 0; p < 8; ++p) {
        int k = kbase + ((p & 4) << 2) + hs * 8 + ((p & 3) << 1);
        f.u[p] = r[k >> 1];
    }
    return f.v;
}

// ---------------------------------------------------------------------------
// Tensor Data Mover: async DMA of one 2048-byte x-tile (512 dwords, 1 row)
// from global into an LDS double buffer. Tracked by TENSORcnt.
// ---------------------------------------------------------------------------
#if __has_builtin(__builtin_amdgcn_tensor_load_to_lds)
#define HAVE_TDM 1
typedef __attribute__((ext_vector_type(4))) unsigned int u32x4;
typedef __attribute__((ext_vector_type(8))) int          i32x8;
typedef __attribute__((ext_vector_type(4))) int          i32x4;

__device__ __forceinline__ void tdm_load_x_tile(unsigned lds_byte_addr,
                                                unsigned long long gaddr) {
    u32x4 g0;
    g0[0] = 1u;                                            // count=1, user mode
    g0[1] = lds_byte_addr;                                 // lds_addr
    g0[2] = (unsigned)(gaddr & 0xffffffffull);             // global_addr[31:0]
    g0[3] = (unsigned)((gaddr >> 32) & 0x1ffffffull)       // global_addr[56:32]
            | (2u << 30);                                  // type = 2 (image)
    i32x8 g1;
    g1[0] = (2 << 16);        // workgroup_mask=0, data_size=2 (4 bytes)
    g1[1] = (512 << 16);      // tensor_dim0[15:0]=512 (atomic_barrier_addr=0)
    g1[2] = (1 << 16);        // tensor_dim0[31:16]=0 | tensor_dim1=1
    g1[3] = (512 << 16);      // tensor_dim1[31:16]=0 | tile_dim0=512
    g1[4] = 1;                // tile_dim1=1, tile_dim2=0
    g1[5] = 512;              // tensor_dim0_stride[31:0]=512
    g1[6] = (512 << 16);      // stride0[47:32]=0 | tensor_dim1_stride[15:0]=512
    g1[7] = 0;                // tensor_dim1_stride[47:16]=0
    i32x4 z4 = {0, 0, 0, 0};
#if __has_include(<hip/amd_detail/amd_gfx1250_TDM.h>)
    // amdgpu-toolchain (clang-23 / therock-10.0 headers): 6-arg builtin
    i32x8 z8 = {0, 0, 0, 0, 0, 0, 0, 0};
    __builtin_amdgcn_tensor_load_to_lds(g0, g1, z4, z4, z8, 0);
#else
    // ROCm 7.2 (clang-22): 5-arg builtin
    __builtin_amdgcn_tensor_load_to_lds(g0, g1, z4, z4, 0);
#endif
}
#endif

// ---------------------------------------------------------------------------
// Kernel 2: fused GRU scan. Grid = 8 blocks (batch tiles of 16 rows),
// block = 768 threads = 24 waves; wave w owns gate-column tile w (16 cols).
// Weight B-fragments (f16) pinned in VGPRs for all 4096 steps; x tiles
// streamed via TDM double-buffer; 6 WMMA / wave / step.
// ---------------------------------------------------------------------------
__global__ void __launch_bounds__(768)
gru_kernel(const _Float16* __restrict__ x2h,
           const float* __restrict__ W_ih, const float* __restrict__ W_hh,
           const float* __restrict__ b_ih, const float* __restrict__ b_hh,
           const float* __restrict__ W_fc, const float* __restrict__ b_fc,
           const float* __restrict__ st11, const float* __restrict__ st12,
           float* __restrict__ out) {
    __shared__ _Float16 sX[2][16 * Isz];   // x2 tile double buffer    (4 KB)
    __shared__ _Float16 sHa[16 * Hsz];     // h (f16), A layout feed   (4 KB)
    __shared__ float    sH [16 * Hsz];     // h (f32 master)           (8 KB)
    __shared__ float    sS  [16 * 256];    // r/z pre-acts (gi+gh)    (16 KB)
    __shared__ float    sGin[16 * Hsz];    // n-gate input path        (8 KB)
    __shared__ float    sGhn[16 * Hsz];    // n-gate hidden path       (8 KB)

    const int tid  = threadIdx.x;
    const int bt   = blockIdx.x;      // batch tile 0..7
    const int wave = tid >> 5;        // 0..23 == gate N-tile
    const int lane = tid & 31;
    const int lm   = lane & 15;
    const int hs   = lane >> 4;

    // ---- one-time: weight B-fragments (B[k][n] = W[n_global][k]) -> VGPRs.
    const int ng = wave * 16 + lm;    // gate column 0..383
    Frag16 whh[4], wih[2];
#pragma unroll
    for (int ks = 0; ks < 4; ++ks)
#pragma unroll
        for (int e = 0; e < 16; ++e)
            whh[ks].h[e] = (_Float16)W_hh[ng * Hsz + ks * 32 + ((e & 8) << 1) + hs * 8 + (e & 7)];
#pragma unroll
    for (int ks = 0; ks < 2; ++ks)
#pragma unroll
        for (int e = 0; e < 16; ++e)
            wih[ks].h[e] = (_Float16)W_ih[ng * Isz + ks * 32 + ((e & 8) << 1) + hs * 8 + (e & 7)];
    const float bi = b_ih[ng];
    const float bh = b_hh[ng];

    // ---- h0 = 0
    for (int idx = tid; idx < 16 * Hsz; idx += 768) {
        sH[idx] = 0.0f;
        sHa[idx] = (_Float16)0.0f;
    }

    const unsigned long long xg_base =
        (unsigned long long)(size_t)(const void*)x2h + (size_t)bt * 2048;

#if defined(HAVE_TDM)
    const unsigned lds_x0 = (unsigned)(size_t)(void*)&sX[0][0];
    // prologue: DMA tile t=0 into buffer 0, wait, publish via barrier
    if (wave == 0) {
        tdm_load_x_tile(lds_x0, xg_base);
        __builtin_amdgcn_s_wait_tensorcnt(0);
    }
#else
    unsigned int* sXu = (unsigned int*)&sX[0][0];
    const unsigned int* xg = (const unsigned int*)x2h;
    if (tid < 512) sXu[tid] = xg[(size_t)bt * 512 + tid];
#endif
    __syncthreads();

    for (int t = 0; t < Tsz; ++t) {
#if defined(HAVE_TDM)
        // kick DMA for t+1 into the other buffer; overlaps with WMMA + gates
        if (wave == 0 && (t + 1) < Tsz)
            tdm_load_x_tile(lds_x0 + (unsigned)(((t + 1) & 1) * 2048),
                            xg_base + (unsigned long long)(t + 1) * 16384ull);
#endif
        const _Float16* hrow = sHa + lm * Hsz;
        const _Float16* xrow = &sX[t & 1][0] + lm * Isz;
        v16h ha0 = load_a_frag(hrow, hs, 0);
        v16h ha1 = load_a_frag(hrow, hs, 32);
        v16h ha2 = load_a_frag(hrow, hs, 64);
        v16h ha3 = load_a_frag(hrow, hs, 96);
        v16h xa0 = load_a_frag(xrow, hs, 0);
        v16h xa1 = load_a_frag(xrow, hs, 32);

        v8f ai = {bi, bi, bi, bi, bi, bi, bi, bi};   // x2_t @ W_ih^T + b_ih
        v8f ah = {bh, bh, bh, bh, bh, bh, bh, bh};   // h    @ W_hh^T + b_hh
        ai = __builtin_amdgcn_wmma_f32_16x16x32_f16(false, xa0, false, wih[0].v, (short)0, ai, false, false);
        ai = __builtin_amdgcn_wmma_f32_16x16x32_f16(false, xa1, false, wih[1].v, (short)0, ai, false, false);
        ah = __builtin_amdgcn_wmma_f32_16x16x32_f16(false, ha0, false, whh[0].v, (short)0, ah, false, false);
        ah = __builtin_amdgcn_wmma_f32_16x16x32_f16(false, ha1, false, whh[1].v, (short)0, ah, false, false);
        ah = __builtin_amdgcn_wmma_f32_16x16x32_f16(false, ha2, false, whh[2].v, (short)0, ah, false, false);
        ah = __builtin_amdgcn_wmma_f32_16x16x32_f16(false, ha3, false, whh[3].v, (short)0, ah, false, false);

        // D layout: VGPR v, lane L -> M = v + 8*(L>=16), N = L&15.
        // r/z tiles (waves 0..15): store summed pre-activation.
        // n tiles (waves 16..23): store both paths (r gates the hidden path).
        if (wave < 16) {
#pragma unroll
            for (int v = 0; v < 8; ++v)
                sS[(v + (hs << 3)) * 256 + ng] = ai[v] + ah[v];
        } else {
#pragma unroll
            for (int v = 0; v < 8; ++v) {
                int m = v + (hs << 3);
                sGin[m * Hsz + (ng - 256)] = ai[v];
                sGhn[m * Hsz + (ng - 256)] = ah[v];
            }
        }
        __syncthreads();

        // gates + state update (fast sigmoid/tanh via v_exp + v_rcp)
        for (int idx = tid; idx < 16 * Hsz; idx += 768) {
            int m = idx >> 7, j = idx & 127;
            float r = __builtin_amdgcn_rcpf(1.0f + __expf(-sS[m * 256 + j]));
            float z = __builtin_amdgcn_rcpf(1.0f + __expf(-sS[m * 256 + 128 + j]));
            float a = sGin[m * Hsz + j] + r * sGhn[m * Hsz + j];
            float q = __expf(-2.0f * a);                       // tanh(a)
            float n = (1.0f - q) * __builtin_amdgcn_rcpf(1.0f + q);
            float hnew = (1.0f - z) * n + z * sH[idx];
            sH[idx]  = hnew;
            sHa[idx] = (_Float16)hnew;
        }

#if defined(HAVE_TDM)
        if (wave == 0 && (t + 1) < Tsz)
            __builtin_amdgcn_s_wait_tensorcnt(0);   // next x tile landed
#else
        __syncthreads();
        if (tid < 512 && (t + 1) < Tsz) {
            unsigned int* sXu2 = (unsigned int*)&sX[(t + 1) & 1][0];
            const unsigned int* xg2 = (const unsigned int*)x2h;
            sXu2[tid] = xg2[(size_t)(t + 1) * 4096 + (size_t)bt * 512 + tid];
        }
#endif
        __syncthreads();
    }

    // ---- epilogue: out = h @ W_fc^T + b_fc, then invert the two EMA trends
    if (tid < 32) {
        int m = tid >> 1, c = tid & 1;
        float acc = b_fc[c];
        const float* hr = sH + m * Hsz;
        const float* wr = W_fc + c * Hsz;
#pragma unroll 8
        for (int j = 0; j < Hsz; ++j) acc += hr[j] * wr[j];
        int b = bt * 16 + m;
        float o = (acc - BETA_C  * st12[b * 2 + c]) * (1.0f / (1.0f - BETA_C));
        o       = (o   - ALPHA_C * st11[b * 2 + c]) * (1.0f / (1.0f - ALPHA_C));
        out[b * Csz + c] = o;
    }
}

// ---------------------------------------------------------------------------
extern "C" void kernel_launch(void* const* d_in, const int* in_sizes, int n_in,
                              void* d_out, int out_size, void* d_ws, size_t ws_size,
                              hipStream_t stream) {
    (void)in_sizes; (void)n_in; (void)out_size; (void)ws_size;
    const float* x    = (const float*)d_in[0];
    const float* W_ih = (const float*)d_in[1];
    const float* W_hh = (const float*)d_in[2];
    const float* b_ih = (const float*)d_in[3];
    const float* b_hh = (const float*)d_in[4];
    const float* W_fc = (const float*)d_in[5];
    const float* b_fc = (const float*)d_in[6];
    float* out = (float*)d_out;

    char* ws = (char*)d_ws;
    _Float16* x2h = (_Float16*)ws;                                   // 67.1 MB
    size_t x2_bytes = (size_t)Tsz * Bsz * Isz * sizeof(_Float16);
    float* st11 = (float*)(ws + x2_bytes);                           // B*2 f32
    float* st12 = st11 + Bsz * 2;                                    // B*2 f32

    ema_kernel<<<(Bsz * Isz) / 256, 256, 0, stream>>>(x, x2h, st11, st12);
    gru_kernel<<<Bsz / 16, 768, 0, stream>>>(x2h, W_ih, W_hh, b_ih, b_hh,
                                             W_fc, b_fc, st11, st12, out);
}
